// SALT_68942815035605
// MI455X (gfx1250) — compile-verified, matrix operand order checked by reading, not run on hardware
//
#include <hip/hip_runtime.h>
#include <hip/hip_bf16.h>
#include <stdint.h>

typedef __attribute__((ext_vector_type(16))) _Float16 v16h;
typedef __attribute__((ext_vector_type(8)))  _Float16 v8h;
typedef __attribute__((ext_vector_type(8)))  float    v8f;

#define NNODE 512
#define TSTEP 8
#define NEDGE 16384
#define BNPAIR (NNODE*NNODE)   // 262144

// ---------------- elementwise convert f32 -> f16 ----------------
__global__ void k_cvt16(const float* __restrict__ in, _Float16* __restrict__ out, int n) {
  int i = blockIdx.x*blockDim.x + threadIdx.x;
  if (i < n) out[i] = (_Float16)in[i];
}

// ---------------- transpose+convert weight: B[K,Nc] -> Bt[Nc,Kpad] f16, zero-pad K ----------------
__global__ void k_transpose16(const float* __restrict__ B, _Float16* __restrict__ Bt,
                              int K, int Nc, int Kpad) {
  int i = blockIdx.x*blockDim.x + threadIdx.x;
  if (i >= Nc*Kpad) return;
  int n = i / Kpad, k = i - n*Kpad;
  Bt[i] = (k < K) ? (_Float16)B[k*Nc + n] : (_Float16)0.f;
}

// ---------------- WMMA GEMM: C = relu(A@B + bias) ----------------
// A: f16 [M,K] row-major.  Bt: f16 [Nc,K] row-major (i.e. B transposed).
// One wave computes a 16x64 output strip: the A fragment is reused across
// 4 N-tiles per K step (4 WMMAs per A fetch) to raise WMMA:VMEM density.
__global__ __launch_bounds__(32) void k_wmma_gemm(
    const _Float16* __restrict__ A, const _Float16* __restrict__ Bt,
    const float* __restrict__ bias, float* __restrict__ Cf32,
    _Float16* __restrict__ Cf16, int M, int K, int Nc, int do_relu) {
  const int lane = threadIdx.x;
  const int grp  = lane >> 4;      // 0: K {0..7,16..23}, 1: K {8..15,24..31}
  const int l    = lane & 15;
  const _Float16* ap  = A  + (size_t)(blockIdx.x*16 + l) * K;
  const _Float16* bp0 = Bt + (size_t)(blockIdx.y*64 +  0 + l) * K;
  const _Float16* bp1 = Bt + (size_t)(blockIdx.y*64 + 16 + l) * K;
  const _Float16* bp2 = Bt + (size_t)(blockIdx.y*64 + 32 + l) * K;
  const _Float16* bp3 = Bt + (size_t)(blockIdx.y*64 + 48 + l) * K;
  v8f acc0 = {}, acc1 = {}, acc2 = {}, acc3 = {};
  for (int k = 0; k < K; k += 32) {
    const int kb = k + grp*8;
    v8h alo = *(const v8h*)(ap + kb);
    v8h ahi = *(const v8h*)(ap + kb + 16);
    v16h a = __builtin_shufflevector(alo, ahi, 0,1,2,3,4,5,6,7,8,9,10,11,12,13,14,15);
    {
      v8h blo = *(const v8h*)(bp0 + kb);
      v8h bhi = *(const v8h*)(bp0 + kb + 16);
      v16h b = __builtin_shufflevector(blo, bhi, 0,1,2,3,4,5,6,7,8,9,10,11,12,13,14,15);
      acc0 = __builtin_amdgcn_wmma_f32_16x16x32_f16(false, a, false, b, (short)0, acc0, false, false);
    }
    {
      v8h blo = *(const v8h*)(bp1 + kb);
      v8h bhi = *(const v8h*)(bp1 + kb + 16);
      v16h b = __builtin_shufflevector(blo, bhi, 0,1,2,3,4,5,6,7,8,9,10,11,12,13,14,15);
      acc1 = __builtin_amdgcn_wmma_f32_16x16x32_f16(false, a, false, b, (short)0, acc1, false, false);
    }
    {
      v8h blo = *(const v8h*)(bp2 + kb);
      v8h bhi = *(const v8h*)(bp2 + kb + 16);
      v16h b = __builtin_shufflevector(blo, bhi, 0,1,2,3,4,5,6,7,8,9,10,11,12,13,14,15);
      acc2 = __builtin_amdgcn_wmma_f32_16x16x32_f16(false, a, false, b, (short)0, acc2, false, false);
    }
    {
      v8h blo = *(const v8h*)(bp3 + kb);
      v8h bhi = *(const v8h*)(bp3 + kb + 16);
      v16h b = __builtin_shufflevector(blo, bhi, 0,1,2,3,4,5,6,7,8,9,10,11,12,13,14,15);
      acc3 = __builtin_amdgcn_wmma_f32_16x16x32_f16(false, a, false, b, (short)0, acc3, false, false);
    }
  }
  // epilogue: D layout per ISA -- lane group selects M rows 0..7 vs 8..15,
  // lane%16 = N within the 16-wide subtile.
  v8f accs[4] = {acc0, acc1, acc2, acc3};
#pragma unroll
  for (int sub = 0; sub < 4; ++sub) {
    const int col = blockIdx.y*64 + sub*16 + l;
    const float bv = bias ? bias[col] : 0.f;
#pragma unroll
    for (int r = 0; r < 8; ++r) {
      int row = blockIdx.x*16 + grp*8 + r;
      float v = accs[sub][r] + bv;
      if (do_relu) v = v > 0.f ? v : 0.f;
      size_t o = (size_t)row*Nc + col;
      if (Cf32) Cf32[o] = v;
      if (Cf16) Cf16[o] = (_Float16)v;
    }
  }
}

// ---------------- emb = concat(proj64, noise16, zeropad16) as f16 [4096,96] ----------------
__global__ void k_build_emb(const _Float16* __restrict__ proj, const float* __restrict__ noise,
                            _Float16* __restrict__ emb, int total) {
  int i = blockIdx.x*blockDim.x + threadIdx.x;
  if (i >= total) return;
  int r = i / 96, j = i - r*96;
  _Float16 v;
  if (j < 64)      v = proj[r*64 + j];
  else if (j < 80) v = (_Float16)noise[r*16 + (j - 64)];
  else             v = (_Float16)0.f;
  emb[i] = v;
}

// ---------------- agg[dst] += ew * hh[src]  (ei pre-offset to timestep) ----------------
__global__ void k_scatter_agg(const int* __restrict__ ei, const float* __restrict__ ew,
                              const float* __restrict__ hh, float* __restrict__ agg) {
  int i = blockIdx.x*blockDim.x + threadIdx.x;   // NEDGE*128 exact
  int e = i >> 7, f = i & 127;
  int dst = ei[e];
  int src = ei[NEDGE + e];
  atomicAdd(&agg[dst*128 + f], ew[e] * hh[src*128 + f]);
}

// ---------------- x2 = f16(hh + agg) ----------------
__global__ void k_add_cvt(const float* __restrict__ hh, const float* __restrict__ agg,
                          _Float16* __restrict__ x2, int n) {
  int i = blockIdx.x*blockDim.x + threadIdx.x;
  if (i < n) x2[i] = (_Float16)(hh[i] + agg[i]);
}

// ---------------- ef[e,f] = f16(h2[ei0[e],f] * h2[ei1[e],f]) ----------------
__global__ void k_gather_prod(const int* __restrict__ ei, const float* __restrict__ h2,
                              _Float16* __restrict__ ef) {
  int i = blockIdx.x*blockDim.x + threadIdx.x;   // NEDGE*128 exact
  int e = i >> 7, f = i & 127;
  ef[i] = (_Float16)(h2[ei[e]*128 + f] * h2[ei[NEDGE + e]*128 + f]);
}

// ---------------- s[e] = g[e,:] . w4 + b4 ----------------
__global__ void k_dot_w4(const float* __restrict__ g, const float* __restrict__ w4,
                         const float* __restrict__ b4, float* __restrict__ s) {
  int e = blockIdx.x*blockDim.x + threadIdx.x;   // NEDGE exact
  const float4* gp = (const float4*)(g + (size_t)e*128);
  const float4* wp = (const float4*)w4;
  float acc = b4[0];
#pragma unroll
  for (int i = 0; i < 32; ++i) {
    float4 gg = gp[i], ww = wp[i];
    acc += gg.x*ww.x + gg.y*ww.y + gg.z*ww.z + gg.w*ww.w;
  }
  s[e] = acc;
}

// ---------------- dense[ei0,ei1] = s[e]  (racy store == JAX .set dup semantics) ----------------
__global__ void k_scatter_dense(const int* __restrict__ ei, const float* __restrict__ s,
                                float* __restrict__ dense_t) {
  int e = blockIdx.x*blockDim.x + threadIdx.x;   // NEDGE exact
  dense_t[ei[e]*NNODE + ei[NEDGE + e]] = s[e];
}

// ---------------- TLSSM scan, one thread per (i,j) pair ----------------
// Key trick: xBC_raw = xt * W_in[0,1+j] is rank-1 => conv_state collapses to
// the last 4 scalar inputs: conv(j) = W_in[1+j] * (xhist . conv_w[j,:]).
__global__ __launch_bounds__(256) void k_tlssm(
    const float* __restrict__ dense, const float* __restrict__ W_in,
    const float* __restrict__ conv_w, const float* __restrict__ conv_b,
    const float* __restrict__ dt_bias, const float* __restrict__ A_log,
    const float* __restrict__ D_skip, const float* __restrict__ W_out,
    float* __restrict__ out) {
  __shared__ float sW[35], scw[132], scb[33], ssc[4];
  const int tid = threadIdx.x;
  if (tid < 35)  sW[tid]  = W_in[tid];
  if (tid < 132) scw[tid] = conv_w[tid];
  if (tid < 33)  scb[tid] = conv_b[tid];
  if (tid == 0) {
    ssc[0] = dt_bias[0];
    ssc[1] = -__expf(A_log[0]);   // A = -exp(A_log)
    ssc[2] = D_skip[0];
    ssc[3] = W_out[0];
  }
  __syncthreads();
  const float dtb = ssc[0], Aneg = ssc[1], Dsk = ssc[2], Wo = ssc[3];

  const int b = blockIdx.x*256 + tid;            // BNPAIR exact
  float xh0 = 0.f, xh1 = 0.f, xh2 = 0.f, xh3 = 0.f;
  float ssm[16];
#pragma unroll
  for (int k = 0; k < 16; ++k) ssm[k] = 0.f;
  float ylast = 0.f;

  for (int t = 0; t < TSTEP; ++t) {
    float xt  = dense[t*BNPAIR + b];
    float z   = xt * sW[0];
    float dtr = xt * sW[34] + dtb;
    xh0 = xh1; xh1 = xh2; xh2 = xh3; xh3 = xt;
    float xBC[33];
#pragma unroll
    for (int j = 0; j < 33; ++j) {
      float cv = xh0*scw[4*j] + xh1*scw[4*j+1] + xh2*scw[4*j+2] + xh3*scw[4*j+3];
      float p  = sW[1+j]*cv + scb[j];
      xBC[j]   = p / (1.f + __expf(-p));         // silu
    }
    float x_ = xBC[0];
    float dt = (dtr > 20.f) ? dtr : log1pf(__expf(dtr));   // softplus
    float dA = __expf(dt * Aneg);
    float dx = dt * x_;
    float y  = 0.f;
#pragma unroll
    for (int k = 0; k < 16; ++k) {
      ssm[k] = ssm[k]*dA + dx*xBC[1+k];          // *B
      y     += ssm[k]*xBC[17+k];                 // .C
    }
    y += Dsk * x_;
    y *= z / (1.f + __expf(-z));                 // * silu(z)
    ylast = y * Wo;
  }
  out[b] = ylast;                                // only last step is emitted (PRED=1)
}

// =======================================================================
extern "C" void kernel_launch(void* const* d_in, const int* in_sizes, int n_in,
                              void* d_out, int out_size, void* d_ws, size_t ws_size,
                              hipStream_t stream) {
  (void)in_sizes; (void)n_in; (void)out_size; (void)ws_size;

  const int*   ei      = (const int*)  d_in[0];   // [T,2,E]
  const float* ew      = (const float*)d_in[1];   // [T,E]
  const float* feats   = (const float*)d_in[2];   // [T,N,128]
  const float* noise   = (const float*)d_in[3];   // [T,N,16]
  const float* Wp1     = (const float*)d_in[4];
  const float* bp1     = (const float*)d_in[5];
  const float* Wp2     = (const float*)d_in[6];
  const float* bp2     = (const float*)d_in[7];
  const float* W1      = (const float*)d_in[8];
  const float* b1      = (const float*)d_in[9];
  const float* W2      = (const float*)d_in[10];
  const float* b2      = (const float*)d_in[11];
  const float* W3      = (const float*)d_in[12];
  const float* b3      = (const float*)d_in[13];
  const float* w4      = (const float*)d_in[14];
  const float* b4      = (const float*)d_in[15];
  const float* W_in    = (const float*)d_in[16];
  const float* conv_w  = (const float*)d_in[17];
  const float* conv_b  = (const float*)d_in[18];
  const float* dt_bias = (const float*)d_in[19];
  const float* A_log   = (const float*)d_in[20];
  const float* D_skip  = (const float*)d_in[21];
  const float* W_out   = (const float*)d_in[22];
  float* out = (float*)d_out;

  // ---- workspace bump allocator (deterministic layout, 256B aligned) ----
  char* ws = (char*)d_ws;
  size_t off = 0;
  auto alloc = [&](size_t bytes) -> void* {
    off = (off + 255) & ~(size_t)255;
    void* p = ws + off;
    off += bytes;
    return p;
  };
  const int TN = TSTEP * NNODE;                   // 4096
  _Float16* feats16 = (_Float16*)alloc((size_t)TN*128*2);
  _Float16* Wp1t    = (_Float16*)alloc(256*128*2);
  _Float16* Wp2t    = (_Float16*)alloc(64*256*2);
  _Float16* h1_16   = (_Float16*)alloc((size_t)TN*256*2);
  _Float16* proj16  = (_Float16*)alloc((size_t)TN*64*2);
  _Float16* emb16   = (_Float16*)alloc((size_t)TN*96*2);
  _Float16* W1t     = (_Float16*)alloc(128*96*2);
  _Float16* W2t     = (_Float16*)alloc(128*128*2);
  _Float16* W3t     = (_Float16*)alloc(128*128*2);
  float*    hh32    = (float*)   alloc((size_t)NNODE*128*4);
  float*    agg32   = (float*)   alloc((size_t)NNODE*128*4);
  _Float16* x2_16   = (_Float16*)alloc((size_t)NNODE*128*2);
  float*    h2_32   = (float*)   alloc((size_t)NNODE*128*4);
  _Float16* ef16    = (_Float16*)alloc((size_t)NEDGE*128*2);
  float*    g32     = (float*)   alloc((size_t)NEDGE*128*4);
  float*    s32     = (float*)   alloc((size_t)NEDGE*4);
  float*    dense   = (float*)   alloc((size_t)TSTEP*BNPAIR*4);

  const dim3 b256(256), bwave(32);

  // ---- precision staging: activations + transposed weights to f16 ----
  k_cvt16<<<dim3((TN*128)/256), b256, 0, stream>>>(feats, feats16, TN*128);
  k_transpose16<<<dim3((256*128)/256), b256, 0, stream>>>(Wp1, Wp1t, 128, 256, 128);
  k_transpose16<<<dim3((64*256)/256),  b256, 0, stream>>>(Wp2, Wp2t, 256, 64, 256);
  k_transpose16<<<dim3((128*96)/256),  b256, 0, stream>>>(W1,  W1t,  80, 128, 96);
  k_transpose16<<<dim3((128*128)/256), b256, 0, stream>>>(W2,  W2t, 128, 128, 128);
  k_transpose16<<<dim3((128*128)/256), b256, 0, stream>>>(W3,  W3t, 128, 128, 128);

  // ---- feat_proj: two WMMA GEMMs with ReLU ----
  k_wmma_gemm<<<dim3(TN/16, 256/64), bwave, 0, stream>>>(
      feats16, Wp1t, bp1, nullptr, h1_16, TN, 128, 256, 1);
  k_wmma_gemm<<<dim3(TN/16, 64/64),  bwave, 0, stream>>>(
      h1_16, Wp2t, bp2, nullptr, proj16, TN, 256, 64, 1);
  k_build_emb<<<dim3((TN*96)/256), b256, 0, stream>>>(proj16, noise, emb16, TN*96);

  hipMemsetAsync(dense, 0, (size_t)TSTEP*BNPAIR*4, stream);

  // ---- WLE per timestep ----
  for (int t = 0; t < TSTEP; ++t) {
    const int*   ei_t = ei + (size_t)t*2*NEDGE;
    const float* ew_t = ew + (size_t)t*NEDGE;
    // hh = relu(emb @ W1 + b1)   [512,96]x[96,128]
    k_wmma_gemm<<<dim3(NNODE/16, 128/64), bwave, 0, stream>>>(
        emb16 + (size_t)t*NNODE*96, W1t, b1, hh32, nullptr, NNODE, 96, 128, 1);
    hipMemsetAsync(agg32, 0, (size_t)NNODE*128*4, stream);
    k_scatter_agg<<<dim3((NEDGE*128)/256), b256, 0, stream>>>(ei_t, ew_t, hh32, agg32);
    k_add_cvt<<<dim3((NNODE*128)/256), b256, 0, stream>>>(hh32, agg32, x2_16, NNODE*128);
    // h2 = relu((hh+agg) @ W2 + b2)   [512,128]x[128,128]
    k_wmma_gemm<<<dim3(NNODE/16, 128/64), bwave, 0, stream>>>(
        x2_16, W2t, b2, h2_32, nullptr, NNODE, 128, 128, 1);
    k_gather_prod<<<dim3((NEDGE*128)/256), b256, 0, stream>>>(ei_t, h2_32, ef16);
    // g = relu(ef @ W3 + b3)   [16384,128]x[128,128]  (dominant GEMM)
    k_wmma_gemm<<<dim3(NEDGE/16, 128/64), bwave, 0, stream>>>(
        ef16, W3t, b3, g32, nullptr, NEDGE, 128, 128, 1);
    k_dot_w4<<<dim3(NEDGE/256), b256, 0, stream>>>(g32, w4, b4, s32);
    k_scatter_dense<<<dim3(NEDGE/256), b256, 0, stream>>>(ei_t, s32, dense + (size_t)t*BNPAIR);
  }

  // ---- TLSSM scan over the window; writes the final [512,512] map ----
  k_tlssm<<<dim3(BNPAIR/256), b256, 0, stream>>>(
      dense, W_in, conv_w, conv_b, dt_bias, A_log, D_skip, W_out, out);
}
